// GraphAttentionLayer_83330955477595
// MI455X (gfx1250) — matrix operand order repaired and checked by reading, not compiled, not run on hardware
//
#include <hip/hip_runtime.h>

// GAT layer: N=4096, IN=256, OUT=128, EDGE=5, slope=0.2, NEG=-9e15
#define NN 4096
#define IND 256
#define OUTD 128
#define SLOPE 0.2f
#define NEGV -9000000000000000.0f

typedef __attribute__((ext_vector_type(16))) __bf16 v16bf;
typedef __attribute__((ext_vector_type(8)))  float  v8f;

union FragU { v16bf bf; uint4 q[2]; };

static __device__ __forceinline__ unsigned short f2bf(float x) {
  unsigned int u = __builtin_bit_cast(unsigned int, x);
  u += 0x7FFFu + ((u >> 16) & 1u);          // round-to-nearest-even
  return (unsigned short)(u >> 16);
}

static __device__ __forceinline__ float wave_max32(float v) {
#pragma unroll
  for (int off = 16; off > 0; off >>= 1)
    v = fmaxf(v, __shfl_xor(v, off, 32));
  return v;
}
static __device__ __forceinline__ float wave_sum32(float v) {
#pragma unroll
  for (int off = 16; off > 0; off >>= 1)
    v += __shfl_xor(v, off, 32);
  return v;
}

// ---------------- Kernel 1: Wa_i = W @ a_i, Wa_j = W @ a_j ----------------
__global__ __launch_bounds__(256) void gat_prep(const float* __restrict__ W,
                                                const float* __restrict__ a,
                                                float* __restrict__ Wai,
                                                float* __restrict__ Waj) {
  int k = threadIdx.x;  // 0..255
  float si = 0.f, sj = 0.f;
  for (int n = 0; n < OUTD; ++n) {
    float wv = W[k * OUTD + n];
    si += wv * a[n];              // a_i = a[0:128]
    sj += wv * a[OUTD + 5 + n];   // a_j = a[133:261]
  }
  Wai[k] = si;
  Waj[k] = sj;
}

// ---------------- Kernel 2: s_i[i], s_j[i] = lstm_out[i,:] . Wa_{i,j} -----
__global__ __launch_bounds__(256) void gat_sisj(const float* __restrict__ lstm,
                                                const float* __restrict__ Wai,
                                                const float* __restrict__ Waj,
                                                float* __restrict__ s_i,
                                                float* __restrict__ s_j) {
  int w = threadIdx.x >> 5, lane = threadIdx.x & 31;
  int row = blockIdx.x * 8 + w;
  float ai = 0.f, aj = 0.f;
  for (int k = lane; k < IND; k += 32) {
    float x = lstm[(size_t)row * IND + k];
    ai += x * Wai[k];
    aj += x * Waj[k];
  }
  ai = wave_sum32(ai);
  aj = wave_sum32(aj);
  if (lane == 0) { s_i[row] = ai; s_j[row] = aj; }
}

// ---------------- Kernel 3: whT (bf16, [128][4096]) = (lstm @ W)^T --------
__global__ __launch_bounds__(32) void gat_whT(const float* __restrict__ lstm,
                                              const float* __restrict__ W,
                                              unsigned short* __restrict__ whT) {
  __shared__ __align__(16) unsigned short At[16][32];  // [m][k] bf16
  __shared__ __align__(16) unsigned short Bt[16][32];  // [n][k] bf16 (W^T tile)
  const int lane = threadIdx.x;
  const int n0 = blockIdx.x * 16;  // 0..7 tiles of OUTD
  const int m0 = blockIdx.y * 16;  // 0..255 tiles of NN

  v8f c;
#pragma unroll
  for (int q = 0; q < 8; ++q) c[q] = 0.f;

  for (int kt = 0; kt < IND / 32; ++kt) {
    const int k0 = kt * 32;
#pragma unroll
    for (int r = 0; r < 16; ++r) {
      At[r][lane] = f2bf(lstm[(size_t)(m0 + r) * IND + k0 + lane]);
      Bt[r][lane] = f2bf(W[(size_t)(k0 + lane) * OUTD + n0 + r]);
    }
    __syncthreads();
    FragU af, bfg;
    const char* ab = (const char*)&At[0][0] + (lane & 15) * 64 + (lane >> 4) * 32;
    af.q[0] = *(const uint4*)(ab);
    af.q[1] = *(const uint4*)(ab + 16);
    const char* bb = (const char*)&Bt[0][0] + (lane & 15) * 64 + (lane >> 4) * 32;
    bfg.q[0] = *(const uint4*)(bb);
    bfg.q[1] = *(const uint4*)(bb + 16);
    c = __builtin_amdgcn_wmma_f32_16x16x32_bf16(false, af.bf, false, bfg.bf,
                                                (short)0, c, false, false);
    __syncthreads();
  }
  // C layout: VGPR v, lanes<16 -> row M=v, lanes>=16 -> row M=v+8; col N=lane%16
  const int h = lane >> 4;
  union { unsigned short s[8]; uint4 q; } st;
#pragma unroll
  for (int v = 0; v < 8; ++v) st.s[v] = f2bf(c[v]);
  *(uint4*)(whT + (size_t)(n0 + (lane & 15)) * NN + m0 + 8 * h) = st.q;
}

// ---------------- Kernel 4: fused scores + softmax + attention @ wh -------
// Block = 256 threads (8 waves), one block per 16-row tile.
__global__ __launch_bounds__(256) void gat_attn(const float* __restrict__ edges,
                                                const unsigned short* __restrict__ whT,
                                                const float* __restrict__ s_i,
                                                const float* __restrict__ s_j,
                                                const float* __restrict__ a,
                                                float* __restrict__ out) {
  __shared__ __align__(16) unsigned short Pt[8][16][32];   // per-wave P tiles (bf16)
  __shared__ float accw[8][16][128];                       // per-wave scaled accumulators
  __shared__ float mred[8][16];
  __shared__ float lred[8][16];

  const int tid = threadIdx.x;
  const int w = tid >> 5, lane = tid & 31;
  const bool hi = lane >= 16;
  const int i0 = blockIdx.x * 16;

  const float ae0 = a[128], ae1 = a[129], ae2 = a[130], ae3 = a[131], ae4 = a[132];

  float si[16];
#pragma unroll
  for (int r = 0; r < 16; ++r) si[r] = s_i[i0 + r];

  float m[16], l[16];
#pragma unroll
  for (int r = 0; r < 16; ++r) { m[r] = -3.0e38f; l[r] = 0.f; }
  v8f acc[8];
#pragma unroll
  for (int t = 0; t < 8; ++t)
#pragma unroll
    for (int q = 0; q < 8; ++q) acc[t][q] = 0.f;

  for (int jt = w; jt < NN / 32; jt += 8) {
    const int j0 = jt * 32;
    // next tile this wave will touch (clamped so prefetch stays in-bounds)
    const int jn0 = (jt + 8 < NN / 32) ? (jt + 8) * 32 : j0;
    const float sj = s_j[j0 + lane];
    float scale[16];
#pragma unroll
    for (int r = 0; r < 16; ++r) {
      const size_t eidx = ((size_t)(i0 + r) * NN + (size_t)(j0 + lane)) * 5u;
      const float* e = edges + eidx;
      // streamed once: non-temporal so L2 stays reserved for whT
      float e0 = __builtin_nontemporal_load(e + 0);
      float e1 = __builtin_nontemporal_load(e + 1);
      float e2 = __builtin_nontemporal_load(e + 2);
      float e3 = __builtin_nontemporal_load(e + 3);
      float e4 = __builtin_nontemporal_load(e + 4);
      // prefetch the same row of the next j-tile (stride 80KB, per-lane addr)
      const float* ep = edges + ((size_t)(i0 + r) * NN + (size_t)(jn0 + lane)) * 5u;
      __builtin_prefetch(ep, 0, 0);
      float se = e0 * ae0 + e1 * ae1 + e2 * ae2 + e3 * ae3 + e4 * ae4;
      float sc = si[r] + se + sj;
      sc = sc > 0.f ? sc : SLOPE * sc;                 // leaky_relu
      if (i0 + r == j0 + lane) sc = NEGV;              // diagonal mask
      float rowmax = wave_max32(sc);
      float mnew = fmaxf(m[r], rowmax);
      float corr = __expf(m[r] - mnew);
      float p = __expf(sc - mnew);
      float rowsum = wave_sum32(p);
      l[r] = l[r] * corr + rowsum;
      m[r] = mnew;
      scale[r] = corr;
      Pt[w][r][lane] = f2bf(p);
    }
    // rescale accumulator rows: VGPR v <-> row v (lanes<16) / v+8 (lanes>=16)
#pragma unroll
    for (int v = 0; v < 8; ++v) {
      float cv = hi ? scale[v + 8] : scale[v];
#pragma unroll
      for (int t = 0; t < 8; ++t) acc[t][v] *= cv;
    }
    // A fragment (16x32 bf16) from LDS, contiguous K (k = 16*h + s)
    FragU af;
    const char* ab = (const char*)&Pt[w][0][0] + (lane & 15) * 64 + (lane >> 4) * 32;
    af.q[0] = *(const uint4*)(ab);
    af.q[1] = *(const uint4*)(ab + 16);
    // 8 B fragments (32x16 bf16) straight from whT (contiguous K along rows)
#pragma unroll
    for (int t = 0; t < 8; ++t) {
      FragU bfg;
      const unsigned short* bb =
          whT + (size_t)(t * 16 + (lane & 15)) * NN + j0 + (lane >> 4) * 16;
      bfg.q[0] = *(const uint4*)(bb);
      bfg.q[1] = *(const uint4*)(bb + 8);
      acc[t] = __builtin_amdgcn_wmma_f32_16x16x32_bf16(false, af.bf, false, bfg.bf,
                                                       (short)0, acc[t], false, false);
    }
  }

  // ---- cross-wave combine (deterministic, no atomics) ----
  if (lane < 16) mred[w][lane] = m[lane];
  __syncthreads();
  float f[16];
#pragma unroll
  for (int r = 0; r < 16; ++r) {
    float M8 = mred[0][r];
#pragma unroll
    for (int ww = 1; ww < 8; ++ww) M8 = fmaxf(M8, mred[ww][r]);
    f[r] = __expf(m[r] - M8);
  }
  if (lane < 16) lred[w][lane] = l[lane] * f[lane];
#pragma unroll
  for (int v = 0; v < 8; ++v) {
    float fv = hi ? f[v + 8] : f[v];
    int row = v + (hi ? 8 : 0);
#pragma unroll
    for (int t = 0; t < 8; ++t)
      accw[w][row][t * 16 + (lane & 15)] = acc[t][v] * fv;
  }
  __syncthreads();

  const int col = tid & 127;     // 0..127
  const int half = tid >> 7;     // 0 or 1: split 16 rows across 256 threads
#pragma unroll
  for (int rr = 0; rr < 8; ++rr) {
    int r = rr + half * 8;
    float lt = 0.f, val = 0.f;
#pragma unroll
    for (int ww = 0; ww < 8; ++ww) { lt += lred[ww][r]; val += accw[ww][r][col]; }
    __builtin_nontemporal_store(val / lt, &out[(size_t)(i0 + r) * OUTD + col]);
  }
}

extern "C" void kernel_launch(void* const* d_in, const int* in_sizes, int n_in,
                              void* d_out, int out_size, void* d_ws, size_t ws_size,
                              hipStream_t stream) {
  const float* lstm  = (const float*)d_in[1];   // (4096, 256)
  const float* edges = (const float*)d_in[2];   // (4096*4096, 5)
  const float* W     = (const float*)d_in[3];   // (256, 128)
  const float* a     = (const float*)d_in[4];   // (261, 1)
  float* out = (float*)d_out;                   // (4096, 128)

  // workspace layout
  unsigned short* whT = (unsigned short*)d_ws;                 // 128*4096*2 = 1 MB
  float* s_i = (float*)((char*)d_ws + (1 << 20));              // 16 KB
  float* s_j = s_i + NN;                                       // 16 KB
  float* Wai = s_j + NN;                                       // 1 KB
  float* Waj = Wai + IND;                                      // 1 KB

  gat_prep<<<1, 256, 0, stream>>>(W, a, Wai, Waj);
  gat_sisj<<<NN / 8, 256, 0, stream>>>(lstm, Wai, Waj, s_i, s_j);
  gat_whT<<<dim3(OUTD / 16, NN / 16), 32, 0, stream>>>(lstm, W, whT);
  gat_attn<<<NN / 16, 256, 0, stream>>>(edges, whT, s_i, s_j, a, out);
}
